// TransformerBlock_38817914421287
// MI455X (gfx1250) — matrix operand order repaired and checked by reading, not compiled, not run on hardware
//
#include <hip/hip_runtime.h>

// ---------------------------------------------------------------------------
// Transformer block for MI455X (gfx1250, wave32, WMMA).
// - All matmuls: v_wmma_f32_16x16x32_f16 (f16 operands, f32 accumulate).
// - Weights pre-transposed+converted to f16 once; activations f16 end-to-end.
// - LDS staging uses global_load_async_to_lds_b128 (ASYNCcnt) with ping-pong
//   double buffering in the GEMM.
// ---------------------------------------------------------------------------

#define USE_ASYNC 1

typedef __attribute__((ext_vector_type(16))) _Float16 v16h;
typedef __attribute__((ext_vector_type(8)))  _Float16 v8h;
typedef __attribute__((ext_vector_type(4)))  _Float16 v4h;
typedef __attribute__((ext_vector_type(8)))  float    v8f;

constexpr int Bb = 2, Ss = 2048, Dd = 1024, Hh = 16, HD = 64, FFd = 4096;
constexpr int Mtot = Bb * Ss;          // 4096 rows

__device__ inline v8f vzero8() {
    v8f z;
#pragma unroll
    for (int i = 0; i < 8; ++i) z[i] = 0.0f;
    return z;
}

__device__ inline v16h ldpair(const _Float16* p0, const _Float16* p1) {
    v8h a = *(const v8h*)p0;
    v8h b = *(const v8h*)p1;
    v16h r;
#pragma unroll
    for (int i = 0; i < 8; ++i) { r[i] = a[i]; r[i + 8] = b[i]; }
    return r;
}

__device__ inline v8f wmma16(v16h a, v16h b, v8f c) {
    return __builtin_amdgcn_wmma_f32_16x16x32_f16(
        false, a, false, b, (short)0, c, false, false);
}

// 16-byte global->LDS stage. Async engine path (ASYNCcnt) or plain copy.
__device__ inline void stage_b128(_Float16* lp, const _Float16* base, unsigned goff) {
#if USE_ASYNC
    asm volatile("global_load_async_to_lds_b128 %0, %1, %2"
                 :: "v"((unsigned)(uintptr_t)lp), "v"(goff), "s"(base)
                 : "memory");
#else
    *(v8h*)lp = *(const v8h*)((const char*)base + goff);
#endif
}
__device__ inline void stage_fence() {
#if USE_ASYNC
    asm volatile("s_wait_asynccnt 0x0" ::: "memory");
#endif
}

// ---------------------------------------------------------------------------
// One-time prep kernels (bandwidth-trivial vs the GEMM compute).
// ---------------------------------------------------------------------------
__global__ __launch_bounds__(256) void f32_to_f16_kernel(
    const float* __restrict__ in, _Float16* __restrict__ out)
{
    int i = (blockIdx.x * 256 + threadIdx.x) * 4;
    float4 d = *(const float4*)(in + i);
    v4h h;
    h[0] = (_Float16)d.x; h[1] = (_Float16)d.y;
    h[2] = (_Float16)d.z; h[3] = (_Float16)d.w;
    *(v4h*)(out + i) = h;
}

// Wt[N][K] (f16) = transpose(W[K][N] f32)
__global__ __launch_bounds__(256) void transpose_f16_kernel(
    const float* __restrict__ W, _Float16* __restrict__ Wt, int K, int N)
{
    __shared__ _Float16 t[32][33];
    const int tid = threadIdx.x;
    const int n0 = blockIdx.x * 32, k0 = blockIdx.y * 32;
#pragma unroll
    for (int i = 0; i < 4; ++i) {
        int idx = tid + i * 256;
        int r = idx >> 5, c = idx & 31;
        t[r][c] = (_Float16)W[(size_t)(k0 + r) * N + n0 + c];
    }
    __syncthreads();
#pragma unroll
    for (int i = 0; i < 4; ++i) {
        int idx = tid + i * 256;
        int r = idx >> 5, c = idx & 31;
        Wt[(size_t)(n0 + r) * K + k0 + c] = t[c][r];
    }
}

// ---------------------------------------------------------------------------
// Tiled WMMA GEMM: out = epi(A[M,K]f16 * Wt[N,K]^T f16 + bias (+resid f32))
//   MODE 0: f32 row-major (+resid)      MODE 1: f16 per-head [B,H,S,64]
//   MODE 2: f16 row-major ReLU          MODE 3: f16 V-transposed [B,H,64,S]
// 128x64 block tile, 256 thr = 8 waves (each 32x32), K-step 32,
// ping-pong LDS double buffer filled by async copies.
// ---------------------------------------------------------------------------
template <int MODE, bool RESID>
__global__ __launch_bounds__(256) void gemm_wmma(
    const _Float16* __restrict__ A, const _Float16* __restrict__ Wt,
    const float* __restrict__ bias, const float* __restrict__ resid,
    void* __restrict__ outp, int N, int K)
{
    __shared__ __align__(16) _Float16 As[2][128][32];  // 16 KB
    __shared__ __align__(16) _Float16 Bs[2][64][32];   //  8 KB

    const int tid  = threadIdx.x;
    const int lane = tid & 31;
    const int wave = tid >> 5;
    const int block_m = blockIdx.y * 128;
    const int block_n = blockIdx.x * 64;
    const int moff = (wave & 3) * 32;
    const int noff = (wave >> 2) * 32;
    const int m16  = lane & 15;
    const int aoff = (lane < 16) ? 0 : 8;
    const int boff = (lane < 16) ? 0 : 8;

    // Per-thread constant staging coordinates.
    const int ar0 = tid >> 2,            ac0 = (tid & 3) * 8;        // A chunk 0
    const int ar1 = (tid + 256) >> 2,    ac1 = ((tid + 256) & 3) * 8;// A chunk 1
    const int bn  = tid >> 2,            bc  = (tid & 3) * 8;        // B chunk
    const unsigned agoff0 = (unsigned)(ar0 * K + ac0) * 2u;
    const unsigned agoff1 = (unsigned)(ar1 * K + ac1) * 2u;
    const unsigned bgoff  = (unsigned)(bn  * K + bc ) * 2u;

    v8f acc[2][2];
#pragma unroll
    for (int mt = 0; mt < 2; ++mt)
#pragma unroll
        for (int nt = 0; nt < 2; ++nt) acc[mt][nt] = vzero8();

    // Prologue: stage tile 0 into buffer 0.
    {
        const _Float16* Ab = A  + (size_t)block_m * K;
        const _Float16* Bb_ = Wt + (size_t)block_n * K;
        stage_b128(&As[0][ar0][ac0], Ab, agoff0);
        stage_b128(&As[0][ar1][ac1], Ab, agoff1);
        stage_b128(&Bs[0][bn][bc],   Bb_, bgoff);
    }
    stage_fence();
    __syncthreads();

    int p = 0;
    for (int k0 = 0; k0 < K; k0 += 32) {
        // Stage next tile into the other buffer (overlaps with WMMA below).
        if (k0 + 32 < K) {
            const _Float16* Ab = A  + (size_t)block_m * K + (k0 + 32);
            const _Float16* Bb_ = Wt + (size_t)block_n * K + (k0 + 32);
            stage_b128(&As[p ^ 1][ar0][ac0], Ab, agoff0);
            stage_b128(&As[p ^ 1][ar1][ac1], Ab, agoff1);
            stage_b128(&Bs[p ^ 1][bn][bc],   Bb_, bgoff);
        }
        // Compute on current buffer.
        v16h af[2], bf[2];
#pragma unroll
        for (int mt = 0; mt < 2; ++mt) {
            const _Float16* ar = &As[p][moff + mt * 16 + m16][0];
            af[mt] = ldpair(ar + aoff, ar + 16 + aoff);
        }
#pragma unroll
        for (int nt = 0; nt < 2; ++nt) {
            const _Float16* br = &Bs[p][noff + nt * 16 + m16][0];
            bf[nt] = ldpair(br + boff, br + boff + 16);
        }
#pragma unroll
        for (int mt = 0; mt < 2; ++mt)
#pragma unroll
            for (int nt = 0; nt < 2; ++nt)
                acc[mt][nt] = wmma16(af[mt], bf[nt], acc[mt][nt]);

        stage_fence();
        __syncthreads();
        p ^= 1;
    }

    // Epilogue.
#pragma unroll
    for (int mt = 0; mt < 2; ++mt) {
#pragma unroll
        for (int nt = 0; nt < 2; ++nt) {
            int col = block_n + noff + nt * 16 + m16;
            float bcol = bias[col];
            int rbase = block_m + moff + mt * 16 + ((lane < 16) ? 0 : 8);
#pragma unroll
            for (int r = 0; r < 8; ++r) {
                int row = rbase + r;
                float vv = acc[mt][nt][r] + bcol;
                if constexpr (RESID) vv += resid[(size_t)row * N + col];
                if constexpr (MODE == 0) {
                    ((float*)outp)[(size_t)row * N + col] = vv;
                } else if constexpr (MODE == 2) {
                    vv = vv > 0.0f ? vv : 0.0f;
                    ((_Float16*)outp)[(size_t)row * N + col] = (_Float16)vv;
                } else if constexpr (MODE == 1) {
                    int b = row / Ss, s = row % Ss;
                    int h = col >> 6, dd = col & 63;
                    ((_Float16*)outp)[(((size_t)(b * Hh + h)) * Ss + s) * 64 + dd] =
                        (_Float16)vv;
                } else { // MODE 3: V transposed [bh][64][S]
                    int b = row / Ss, s = row % Ss;
                    int h = col >> 6, dd = col & 63;
                    ((_Float16*)outp)[((size_t)(b * Hh + h) * 64 + dd) * Ss + s] =
                        (_Float16)vv;
                }
            }
        }
    }
}

// ---------------------------------------------------------------------------
// Flash attention: 128 thr = 4 waves, each wave 16 query rows; 64-key chunks
// staged via async copies (V already transposed in memory); online softmax.
// Output written as f16 row-major [B*S, D] (A-operand of the O-projection).
// ---------------------------------------------------------------------------
__global__ __launch_bounds__(128) void attn_kernel(
    const _Float16* __restrict__ q, const _Float16* __restrict__ k,
    const _Float16* __restrict__ vt, const int* __restrict__ mask,
    _Float16* __restrict__ ao)
{
    __shared__ __align__(16) _Float16 Ksh[64][64];      // [key][d]  8 KB
    __shared__ __align__(16) _Float16 Vsh[64][64];      // [d][key]  8 KB
    __shared__ __align__(16) _Float16 Psh[4][16][64];   // per-wave P 8 KB

    const int tid  = threadIdx.x;
    const int lane = tid & 31;
    const int wave = tid >> 5;
    const int bh   = blockIdx.y;
    const int b    = bh / Hh;
    const int h    = bh % Hh;
    const int s0   = blockIdx.x * 64 + wave * 16;
    const int m16  = lane & 15;
    const int aoff = (lane < 16) ? 0 : 8;
    const int boff = (lane < 16) ? 0 : 8;

    // Q fragments (16 rows x 64) in registers.
    const _Float16* qrow = q + ((size_t)bh * Ss + s0 + m16) * HD;
    v16h qf[2];
    qf[0] = ldpair(qrow + aoff,      qrow + 16 + aoff);
    qf[1] = ldpair(qrow + 32 + aoff, qrow + 48 + aoff);

    v8f o[4];
#pragma unroll
    for (int t = 0; t < 4; ++t) o[t] = vzero8();
    float mrow[8], lrow[8];
#pragma unroll
    for (int r = 0; r < 8; ++r) { mrow[r] = -1e30f; lrow[r] = 0.0f; }

    const _Float16* kbase = k  + (size_t)bh * Ss * HD;
    const _Float16* vtb   = vt + (size_t)bh * HD * Ss;

    // Per-thread staging coordinates (64x64 tile, 512 x b128, 4 per thread).
    int srow[4], scol[4];
    unsigned kgoff[4], vgoff[4];
#pragma unroll
    for (int i = 0; i < 4; ++i) {
        int idx = tid + i * 128;
        srow[i] = idx >> 3;
        scol[i] = (idx & 7) * 8;
        kgoff[i] = (unsigned)(srow[i] * HD + scol[i]) * 2u;
        vgoff[i] = (unsigned)(srow[i] * Ss + scol[i]) * 2u;
    }

    for (int kc = 0; kc < Ss; kc += 64) {
        __syncthreads();   // everyone done reading previous chunk
        const _Float16* kb = kbase + (size_t)kc * HD;
        const _Float16* vb = vtb + kc;
#pragma unroll
        for (int i = 0; i < 4; ++i) {
            stage_b128(&Ksh[srow[i]][scol[i]], kb, kgoff[i]);
            stage_b128(&Vsh[srow[i]][scol[i]], vb, vgoff[i]);
        }
        stage_fence();
        __syncthreads();

        // ---- scores = Q*K^T, four 16-key subtiles ----
        v8f sc[4];
#pragma unroll
        for (int j = 0; j < 4; ++j) {
            sc[j] = vzero8();
            const _Float16* br = &Ksh[j * 16 + m16][0];
            v16h bf0 = ldpair(br + boff,      br + boff + 16);
            v16h bf1 = ldpair(br + 32 + boff, br + 32 + boff + 16);
            sc[j] = wmma16(qf[0], bf0, sc[j]);
            sc[j] = wmma16(qf[1], bf1, sc[j]);
            int key = kc + j * 16 + m16;
            float madd = (mask[b * Ss + key] == 0) ? -1e9f : 0.0f;
#pragma unroll
            for (int r = 0; r < 8; ++r) sc[j][r] = sc[j][r] * 0.125f + madd;
        }

        // ---- online softmax ----
        float alpha[8];
#pragma unroll
        for (int r = 0; r < 8; ++r) {
            float cm = fmaxf(fmaxf(sc[0][r], sc[1][r]), fmaxf(sc[2][r], sc[3][r]));
#pragma unroll
            for (int mm = 1; mm < 16; mm <<= 1) cm = fmaxf(cm, __shfl_xor(cm, mm, 32));
            float mn = fmaxf(mrow[r], cm);
            alpha[r] = __expf(mrow[r] - mn);
            mrow[r] = mn;
        }
#pragma unroll
        for (int r = 0; r < 8; ++r) {
            float ps = 0.0f;
#pragma unroll
            for (int j = 0; j < 4; ++j) {
                sc[j][r] = __expf(sc[j][r] - mrow[r]);
                ps += sc[j][r];
            }
#pragma unroll
            for (int mm = 1; mm < 16; mm <<= 1) ps += __shfl_xor(ps, mm, 32);
            lrow[r] = lrow[r] * alpha[r] + ps;
#pragma unroll
            for (int t = 0; t < 4; ++t) o[t][r] *= alpha[r];
        }

        // ---- C-layout P -> LDS -> A-layout fragments ----
        int prow = (lane < 16) ? 0 : 8;
#pragma unroll
        for (int r = 0; r < 8; ++r)
#pragma unroll
            for (int j = 0; j < 4; ++j)
                Psh[wave][prow + r][j * 16 + m16] = (_Float16)sc[j][r];
        __syncthreads();
        const _Float16* pr = &Psh[wave][m16][0];
        v16h pf0 = ldpair(pr + aoff,      pr + 16 + aoff);
        v16h pf1 = ldpair(pr + 32 + aoff, pr + 48 + aoff);

        // ---- O += P * V ----
#pragma unroll
        for (int t = 0; t < 4; ++t) {
            const _Float16* vr = &Vsh[t * 16 + m16][0];
            v16h vf0 = ldpair(vr + boff,      vr + boff + 16);
            v16h vf1 = ldpair(vr + 32 + boff, vr + 32 + boff + 16);
            o[t] = wmma16(pf0, vf0, o[t]);
            o[t] = wmma16(pf1, vf1, o[t]);
        }
    }

    // ---- normalize, write f16 [B*S, D] ----
    float invl[8];
#pragma unroll
    for (int r = 0; r < 8; ++r) invl[r] = 1.0f / fmaxf(lrow[r], 1e-20f);
    int rowbase = s0 + ((lane < 16) ? 0 : 8);
#pragma unroll
    for (int t = 0; t < 4; ++t) {
        int col = h * 64 + t * 16 + m16;
#pragma unroll
        for (int r = 0; r < 8; ++r) {
            int s = rowbase + r;
            ao[((size_t)b * Ss + s) * Dd + col] = (_Float16)(o[t][r] * invl[r]);
        }
    }
}

// ---------------------------------------------------------------------------
// In-place LayerNorm (D=1024), one block per row; optional f16 copy out.
// ---------------------------------------------------------------------------
template <bool F16OUT>
__global__ __launch_bounds__(256) void ln_kernel(
    float* __restrict__ x, const float* __restrict__ g,
    const float* __restrict__ be, _Float16* __restrict__ out16)
{
    __shared__ float red[16];
    const int tid = threadIdx.x;
    float* xr = x + (size_t)blockIdx.x * Dd;

    float vals[4];
    float s = 0.0f, s2 = 0.0f;
#pragma unroll
    for (int i = 0; i < 4; ++i) {
        float v = xr[tid + i * 256];
        vals[i] = v; s += v; s2 += v * v;
    }
#pragma unroll
    for (int mm = 1; mm < 32; mm <<= 1) {
        s  += __shfl_xor(s,  mm, 32);
        s2 += __shfl_xor(s2, mm, 32);
    }
    int lane = tid & 31, wave = tid >> 5;
    if (lane == 0) { red[wave] = s; red[8 + wave] = s2; }
    __syncthreads();
    float ts = 0.0f, ts2 = 0.0f;
#pragma unroll
    for (int w = 0; w < 8; ++w) { ts += red[w]; ts2 += red[8 + w]; }
    float mu  = ts / (float)Dd;
    float var = ts2 / (float)Dd - mu * mu;
    float rs  = rsqrtf(var + 1e-5f);
#pragma unroll
    for (int i = 0; i < 4; ++i) {
        int c = tid + i * 256;
        float o = (vals[i] - mu) * rs * g[c] + be[c];
        xr[c] = o;
        if constexpr (F16OUT) out16[(size_t)blockIdx.x * Dd + c] = (_Float16)o;
    }
}

// ---------------------------------------------------------------------------
extern "C" void kernel_launch(void* const* d_in, const int* in_sizes, int n_in,
                              void* d_out, int out_size, void* d_ws, size_t ws_size,
                              hipStream_t stream)
{
    (void)in_sizes; (void)n_in; (void)out_size; (void)ws_size;
    const float* x    = (const float*)d_in[0];
    const int*   mask = (const int*)  d_in[1];
    const float* Wq = (const float*)d_in[2];  const float* bq = (const float*)d_in[3];
    const float* Wk = (const float*)d_in[4];  const float* bk = (const float*)d_in[5];
    const float* Wv = (const float*)d_in[6];  const float* bv = (const float*)d_in[7];
    const float* Wo = (const float*)d_in[8];  const float* bo = (const float*)d_in[9];
    const float* W1 = (const float*)d_in[10]; const float* b1 = (const float*)d_in[11];
    const float* W2 = (const float*)d_in[12]; const float* b2 = (const float*)d_in[13];
    const float* g1 = (const float*)d_in[14]; const float* be1 = (const float*)d_in[15];
    const float* g2 = (const float*)d_in[16]; const float* be2 = (const float*)d_in[17];
    float* out = (float*)d_out;

    // Workspace carve-up (120 MB).
    char* ws = (char*)d_ws;
    const size_t MB = 1ull << 20;
    _Float16* x16 = (_Float16*)(ws +   0 * MB);  // 8 MB
    _Float16* qh  = (_Float16*)(ws +   8 * MB);  // 8 MB
    _Float16* kh  = (_Float16*)(ws +  16 * MB);  // 8 MB
    _Float16* vth = (_Float16*)(ws +  24 * MB);  // 8 MB  [bh][64][S]
    _Float16* ao  = (_Float16*)(ws +  32 * MB);  // 8 MB
    float*    x1  = (float*)   (ws +  40 * MB);  // 16 MB
    _Float16* x1h = (_Float16*)(ws +  56 * MB);  // 8 MB
    _Float16* hb  = (_Float16*)(ws +  64 * MB);  // 32 MB
    _Float16* Wqt = (_Float16*)(ws +  96 * MB);  // 2 MB
    _Float16* Wkt = (_Float16*)(ws +  98 * MB);  // 2 MB
    _Float16* Wvt = (_Float16*)(ws + 100 * MB);  // 2 MB
    _Float16* Wot = (_Float16*)(ws + 102 * MB);  // 2 MB
    _Float16* W1t = (_Float16*)(ws + 104 * MB);  // 8 MB [FF][D]
    _Float16* W2t = (_Float16*)(ws + 112 * MB);  // 8 MB [D][FF]

    // --- one-time prep: convert x, transpose+convert weights ---
    f32_to_f16_kernel<<<Mtot * Dd / 1024, 256, 0, stream>>>(x, x16);
    transpose_f16_kernel<<<dim3(Dd / 32, Dd / 32),  256, 0, stream>>>(Wq, Wqt, Dd, Dd);
    transpose_f16_kernel<<<dim3(Dd / 32, Dd / 32),  256, 0, stream>>>(Wk, Wkt, Dd, Dd);
    transpose_f16_kernel<<<dim3(Dd / 32, Dd / 32),  256, 0, stream>>>(Wv, Wvt, Dd, Dd);
    transpose_f16_kernel<<<dim3(Dd / 32, Dd / 32),  256, 0, stream>>>(Wo, Wot, Dd, Dd);
    transpose_f16_kernel<<<dim3(FFd / 32, Dd / 32), 256, 0, stream>>>(W1, W1t, Dd, FFd);
    transpose_f16_kernel<<<dim3(Dd / 32, FFd / 32), 256, 0, stream>>>(W2, W2t, FFd, Dd);

    dim3 gD(Dd / 64, Mtot / 128);
    dim3 gF(FFd / 64, Mtot / 128);

    // --- Q, K (head layout), V (pre-transposed per head) ---
    gemm_wmma<1, false><<<gD, 256, 0, stream>>>(x16, Wqt, bq, nullptr, qh,  Dd, Dd);
    gemm_wmma<1, false><<<gD, 256, 0, stream>>>(x16, Wkt, bk, nullptr, kh,  Dd, Dd);
    gemm_wmma<3, false><<<gD, 256, 0, stream>>>(x16, Wvt, bv, nullptr, vth, Dd, Dd);

    // --- attention ---
    attn_kernel<<<dim3(Ss / 64, Bb * Hh), 128, 0, stream>>>(qh, kh, vth, mask, ao);

    // --- O-projection + residual(x) -> x1 ; LN1 (also emits f16 copy) ---
    gemm_wmma<0, true><<<gD, 256, 0, stream>>>(ao, Wot, bo, x, x1, Dd, Dd);
    ln_kernel<true><<<Mtot, 256, 0, stream>>>(x1, g1, be1, x1h);

    // --- FFN ---
    gemm_wmma<2, false><<<gF, 256, 0, stream>>>(x1h, W1t, b1, nullptr, hb, FFd, Dd);
    gemm_wmma<0, true><<<gD, 256, 0, stream>>>(hb, W2t, b2, x1, out, Dd, FFd);
    ln_kernel<false><<<Mtot, 256, 0, stream>>>(out, g2, be2, nullptr);
}
